// PointTransformer_12343736009122
// MI455X (gfx1250) — matrix operand order repaired and checked by reading, not compiled, not run on hardware
//
#include <hip/hip_runtime.h>
#include <hip/hip_bf16.h>

#define Bn 4
#define Nn 4096
#define Cn 128
#define Hn 4

typedef __attribute__((ext_vector_type(16))) _Float16 v16h;
typedef __attribute__((ext_vector_type(8)))  _Float16 h8;
typedef __attribute__((ext_vector_type(8)))  float    v8f;

union HV { v16h v; h8 h[2]; };

// ---------------------------------------------------------------------------
// Kernel 1: fcl1 + per-head positional encoding + q/k/v projections.
// 32 points per 128-thread block for weight reuse (each weight row is loaded
// once per block and applied to 32 points). Outputs f16 Q,K (row-major) and
// V transposed [C][N] so attention's B-operand loads are contiguous.
// ---------------------------------------------------------------------------
__global__ __launch_bounds__(128) void prep_kernel(
    const float* __restrict__ x, const float* __restrict__ p,
    const float* __restrict__ fcl1_w, const float* __restrict__ fcl1_b,
    const float* __restrict__ hq_w, const float* __restrict__ hq_b,
    const float* __restrict__ hk_w, const float* __restrict__ hk_b,
    const float* __restrict__ hv_w, const float* __restrict__ hv_b,
    const float* __restrict__ pe1_w, const float* __restrict__ pe1_b,
    const float* __restrict__ pe2_w, const float* __restrict__ pe2_b,
    _Float16* __restrict__ Qh, _Float16* __restrict__ Kh,
    _Float16* __restrict__ Vt)
{
    __shared__ float bufX[32][Cn];   // x, later xh
    __shared__ float bufF[32][Cn];   // xf (persists across heads)
    __shared__ float bufP[32][Cn];   // pe scratch
    __shared__ float pc[32][3];

    const int c   = threadIdx.x;
    const int blk = blockIdx.x;          // 0..511
    const int b   = blk >> 7;            // Nn/32 = 128 point-groups per batch
    const int n0  = (blk & 127) * 32;

    for (int pp = 0; pp < 32; ++pp)
        bufX[pp][c] = x[((size_t)b * Nn + n0 + pp) * Cn + c];
    if (c < 96) { int pp = c / 3, d = c % 3;
        pc[pp][d] = p[((size_t)b * Nn + n0 + pp) * 3 + d]; }
    __syncthreads();

    // xf = x @ fcl1_w.T + fcl1_b
    {
        float acc[32];
        float bb = fcl1_b[c];
        #pragma unroll
        for (int pp = 0; pp < 32; ++pp) acc[pp] = bb;
        const float* w = fcl1_w + (size_t)c * Cn;
        for (int j = 0; j < Cn; ++j) {
            float wv = w[j];
            #pragma unroll
            for (int pp = 0; pp < 32; ++pp) acc[pp] += bufX[pp][j] * wv;
        }
        __syncthreads();                 // all reads of bufX (x) complete
        #pragma unroll
        for (int pp = 0; pp < 32; ++pp) bufF[pp][c] = acc[pp];
    }
    __syncthreads();

    for (int h = 0; h < Hn; ++h) {
        // pe1 = relu(p @ pe1_w.T + pe1_b)   (Conv1d k=1, 3 -> C)
        {
            const float* w = pe1_w + ((size_t)h * Cn + c) * 3;
            float w0 = w[0], w1 = w[1], w2 = w[2];
            float bb = pe1_b[h * Cn + c];
            #pragma unroll
            for (int pp = 0; pp < 32; ++pp) {
                float t = pc[pp][0] * w0 + pc[pp][1] * w1 + pc[pp][2] * w2 + bb;
                bufP[pp][c] = t > 0.f ? t : 0.f;
            }
        }
        __syncthreads();
        // xh = xf + (pe1 @ pe2_w.T + pe2_b)  -> bufX
        {
            float acc[32];
            float bb = pe2_b[h * Cn + c];
            #pragma unroll
            for (int pp = 0; pp < 32; ++pp) acc[pp] = bb;
            const float* w = pe2_w + ((size_t)h * Cn + c) * Cn;
            for (int j = 0; j < Cn; ++j) {
                float wv = w[j];
                #pragma unroll
                for (int pp = 0; pp < 32; ++pp) acc[pp] += bufP[pp][j] * wv;
            }
            #pragma unroll
            for (int pp = 0; pp < 32; ++pp) bufX[pp][c] = bufF[pp][c] + acc[pp];
        }
        __syncthreads();

        const size_t bh = (size_t)(b * Hn + h);
        // Q uses hk_w/hk_b (reference swaps the names), K uses hq_w/hq_b
        {
            float acc[32];
            float bb = hk_b[h * Cn + c];
            #pragma unroll
            for (int pp = 0; pp < 32; ++pp) acc[pp] = bb;
            const float* w = hk_w + ((size_t)h * Cn + c) * Cn;
            for (int j = 0; j < Cn; ++j) {
                float wv = w[j];
                #pragma unroll
                for (int pp = 0; pp < 32; ++pp) acc[pp] += bufX[pp][j] * wv;
            }
            #pragma unroll
            for (int pp = 0; pp < 32; ++pp)
                Qh[(bh * Nn + n0 + pp) * Cn + c] = (_Float16)acc[pp];
        }
        {
            float acc[32];
            float bb = hq_b[h * Cn + c];
            #pragma unroll
            for (int pp = 0; pp < 32; ++pp) acc[pp] = bb;
            const float* w = hq_w + ((size_t)h * Cn + c) * Cn;
            for (int j = 0; j < Cn; ++j) {
                float wv = w[j];
                #pragma unroll
                for (int pp = 0; pp < 32; ++pp) acc[pp] += bufX[pp][j] * wv;
            }
            #pragma unroll
            for (int pp = 0; pp < 32; ++pp)
                Kh[(bh * Nn + n0 + pp) * Cn + c] = (_Float16)acc[pp];
        }
        {
            float acc[32];
            float bb = hv_b[h * Cn + c];
            #pragma unroll
            for (int pp = 0; pp < 32; ++pp) acc[pp] = bb;
            const float* w = hv_w + ((size_t)h * Cn + c) * Cn;
            for (int j = 0; j < Cn; ++j) {
                float wv = w[j];
                #pragma unroll
                for (int pp = 0; pp < 32; ++pp) acc[pp] += bufX[pp][j] * wv;
            }
            #pragma unroll
            for (int pp = 0; pp < 32; ++pp)    // transposed: [C][N]
                Vt[(bh * Cn + c) * Nn + n0 + pp] = (_Float16)acc[pp];
        }
        __syncthreads();
    }
}

// ---------------------------------------------------------------------------
// Kernel 2: flash attention per (b,h).  Wave owns 16 query rows; 8 waves per
// block -> 128-row Q tile.  Per 32-key block: S via 8x v_wmma_f32_16x16x32_f16
// (two 16-key column tiles x 4 K-chunks), online softmax in the C layout,
// P staged through wave-private LDS to re-enter A layout, P.V via 8 WMMAs
// against transposed V.  No S ever hits memory.
// ---------------------------------------------------------------------------
__global__ __launch_bounds__(256) void attn_kernel(
    const _Float16* __restrict__ Qh, const _Float16* __restrict__ Kh,
    const _Float16* __restrict__ Vt, float* __restrict__ Oh)
{
    __shared__ _Float16 plds[8 * 16 * 32];   // per-wave 16x32 P tile

    const int tid  = threadIdx.x;
    const int wave = tid >> 5;
    const int lane = tid & 31;
    const int r    = lane & 15;              // row (A) / column (B) index
    const int hs   = lane >> 4;              // lane-group selector
    const int bh   = blockIdx.x >> 5;        // Nn/128 = 32 q-tiles per (b,h)
    const int tile = blockIdx.x & 31;
    const int qrow0 = tile * 128 + wave * 16;
    const float scale = 0.088388347648318447f;   // 1/sqrt(128)

    // Q in A layout: lane-group 0 holds K 0-15 of each 32-chunk,
    // lane-group 1 holds K 16-31.
    v16h qa[4];
    {
        const _Float16* qrow = Qh + ((size_t)bh * Nn + qrow0 + r) * Cn;
        #pragma unroll
        for (int cc = 0; cc < 4; ++cc) {
            HV t;
            t.h[0] = *(const h8*)(qrow + cc * 32 + hs * 16);
            t.h[1] = *(const h8*)(qrow + cc * 32 + hs * 16 + 8);
            qa[cc] = t.v;
        }
    }

    v8f Oacc[8] = {};
    float mrun[8], lrun[8];
    #pragma unroll
    for (int j = 0; j < 8; ++j) { mrun[j] = -1e30f; lrun[j] = 0.f; }

    _Float16* pw = plds + wave * 512;
    const _Float16* Kbase = Kh + (size_t)bh * Nn * Cn;
    const _Float16* Vbase = Vt + (size_t)bh * Cn * Nn;

    for (int kb = 0; kb < Nn / 32; ++kb) {
        // ---- S = Q * K^T for two 16-key column tiles ----
        v8f s0 = {}, s1 = {};
        {
            // B layout: lane = column (key), K-halves split across lane groups
            const _Float16* k0row = Kbase + (size_t)(kb * 32 + r) * Cn;
            const _Float16* k1row = Kbase + (size_t)(kb * 32 + 16 + r) * Cn;
            #pragma unroll
            for (int cc = 0; cc < 4; ++cc) {
                HV b0, b1;
                b0.h[0] = *(const h8*)(k0row + cc * 32 + hs * 16);
                b0.h[1] = *(const h8*)(k0row + cc * 32 + hs * 16 + 8);
                b1.h[0] = *(const h8*)(k1row + cc * 32 + hs * 16);
                b1.h[1] = *(const h8*)(k1row + cc * 32 + hs * 16 + 8);
                s0 = __builtin_amdgcn_wmma_f32_16x16x32_f16(
                        false, qa[cc], false, b0.v, (short)0, s0, false, false);
                s1 = __builtin_amdgcn_wmma_f32_16x16x32_f16(
                        false, qa[cc], false, b1.v, (short)0, s1, false, false);
            }
        }
        // ---- online softmax directly in the C layout ----
        float alpha[8];
        #pragma unroll
        for (int j = 0; j < 8; ++j) {
            float v0 = s0[j] * scale, v1 = s1[j] * scale;
            float mx = fmaxf(v0, v1);
            #pragma unroll
            for (int off = 1; off < 16; off <<= 1)
                mx = fmaxf(mx, __shfl_xor(mx, off, 32));
            float mnew = fmaxf(mrun[j], mx);
            alpha[j] = __expf(mrun[j] - mnew);
            mrun[j] = mnew;
            float p0 = __expf(v0 - mnew);
            float p1 = __expf(v1 - mnew);
            float rs = p0 + p1;
            #pragma unroll
            for (int off = 1; off < 16; off <<= 1)
                rs += __shfl_xor(rs, off, 32);
            lrun[j] = lrun[j] * alpha[j] + rs;
            // store P: row = j + 8*hs, cols r (keys 0-15) and r+16 (keys 16-31)
            pw[(j + 8 * hs) * 32 + r]      = (_Float16)p0;
            pw[(j + 8 * hs) * 32 + r + 16] = (_Float16)p1;
        }
        // rescale running O by alpha (per-row, replicated across the 16 lanes)
        #pragma unroll
        for (int oc = 0; oc < 8; ++oc)
            #pragma unroll
            for (int j = 0; j < 8; ++j)
                Oacc[oc][j] *= alpha[j];
        __syncthreads();
        // re-read P in A layout
        HV pa;
        pa.h[0] = *(const h8*)(pw + r * 32 + hs * 16);
        pa.h[1] = *(const h8*)(pw + r * 32 + hs * 16 + 8);
        // ---- O += P * V ----  (B from transposed V: contiguous loads)
        #pragma unroll
        for (int oc = 0; oc < 8; ++oc) {
            const _Float16* vrow = Vbase + (size_t)(oc * 16 + r) * Nn + kb * 32;
            HV vb;
            vb.h[0] = *(const h8*)(vrow + hs * 16);
            vb.h[1] = *(const h8*)(vrow + hs * 16 + 8);
            Oacc[oc] = __builtin_amdgcn_wmma_f32_16x16x32_f16(
                    false, pa.v, false, vb.v, (short)0, Oacc[oc], false, false);
        }
        __syncthreads();
    }
    // normalize and write per-head output (f32, row-major)
    #pragma unroll
    for (int oc = 0; oc < 8; ++oc)
        #pragma unroll
        for (int j = 0; j < 8; ++j) {
            int row = qrow0 + j + 8 * hs;
            int ch  = oc * 16 + r;
            Oh[((size_t)bh * Nn + row) * Cn + ch] = Oacc[oc][j] / lrun[j];
        }
}

// ---------------------------------------------------------------------------
// Kernel 3: concat heads -> mh -> fcl2 -> LayerNorm (no affine) -> +residual.
// 16 points per 128-thread block; also emits the p passthrough output.
// ---------------------------------------------------------------------------
__global__ __launch_bounds__(128) void post_kernel(
    const float* __restrict__ Oh,
    const float* __restrict__ mh_w, const float* __restrict__ mh_b,
    const float* __restrict__ fcl2_w, const float* __restrict__ fcl2_b,
    const float* __restrict__ x, const float* __restrict__ p,
    float* __restrict__ out_y, float* __restrict__ out_p)
{
    __shared__ float catL[16][4 * Cn];
    __shared__ float yL[16][Cn];
    __shared__ float muL[16], rsL[16];

    const int c   = threadIdx.x;
    const int blk = blockIdx.x;          // 0..1023
    const int b   = blk >> 8;            // Nn/16 = 256
    const int n0  = (blk & 255) * 16;

    for (int h = 0; h < Hn; ++h)
        for (int pp = 0; pp < 16; ++pp)
            catL[pp][h * Cn + c] =
                Oh[((size_t)(b * Hn + h) * Nn + n0 + pp) * Cn + c];
    __syncthreads();

    // y1 = cat @ mh_w.T + mh_b
    {
        float acc[16];
        float bb = mh_b[c];
        #pragma unroll
        for (int pp = 0; pp < 16; ++pp) acc[pp] = bb;
        const float* w = mh_w + (size_t)c * (4 * Cn);
        for (int j = 0; j < 4 * Cn; ++j) {
            float wv = w[j];
            #pragma unroll
            for (int pp = 0; pp < 16; ++pp) acc[pp] += catL[pp][j] * wv;
        }
        #pragma unroll
        for (int pp = 0; pp < 16; ++pp) yL[pp][c] = acc[pp];
    }
    __syncthreads();
    // y2 = y1 @ fcl2_w.T + fcl2_b   (store into catL[:, 0:128])
    {
        float acc[16];
        float bb = fcl2_b[c];
        #pragma unroll
        for (int pp = 0; pp < 16; ++pp) acc[pp] = bb;
        const float* w = fcl2_w + (size_t)c * Cn;
        for (int j = 0; j < Cn; ++j) {
            float wv = w[j];
            #pragma unroll
            for (int pp = 0; pp < 16; ++pp) acc[pp] += yL[pp][j] * wv;
        }
        #pragma unroll
        for (int pp = 0; pp < 16; ++pp) catL[pp][c] = acc[pp];
    }
    __syncthreads();
    if (c < 16) {
        float s = 0.f, ss = 0.f;
        for (int j = 0; j < Cn; ++j) { float v = catL[c][j]; s += v; ss += v * v; }
        float mu  = s * (1.f / Cn);
        float var = ss * (1.f / Cn) - mu * mu;
        muL[c] = mu;
        rsL[c] = rsqrtf(var + 1e-5f);
    }
    __syncthreads();
    for (int pp = 0; pp < 16; ++pp) {
        size_t row = (size_t)b * Nn + n0 + pp;
        out_y[row * Cn + c] = (catL[pp][c] - muL[pp]) * rsL[pp] + x[row * Cn + c];
    }
    if (c < 48) {
        int pp = c / 3, d = c % 3;
        size_t row = (size_t)b * Nn + n0 + pp;
        out_p[row * 3 + d] = p[row * 3 + d];
    }
}

extern "C" void kernel_launch(void* const* d_in, const int* in_sizes, int n_in,
                              void* d_out, int out_size, void* d_ws, size_t ws_size,
                              hipStream_t stream) {
    const float* x      = (const float*)d_in[0];
    const float* p      = (const float*)d_in[1];
    const float* fcl1_w = (const float*)d_in[2];
    const float* fcl1_b = (const float*)d_in[3];
    const float* hq_w   = (const float*)d_in[4];
    const float* hq_b   = (const float*)d_in[5];
    const float* hk_w   = (const float*)d_in[6];
    const float* hk_b   = (const float*)d_in[7];
    const float* hv_w   = (const float*)d_in[8];
    const float* hv_b   = (const float*)d_in[9];
    const float* pe1_w  = (const float*)d_in[10];
    const float* pe1_b  = (const float*)d_in[11];
    const float* pe2_w  = (const float*)d_in[12];
    const float* pe2_b  = (const float*)d_in[13];
    const float* mh_w   = (const float*)d_in[14];
    const float* mh_b   = (const float*)d_in[15];
    const float* fcl2_w = (const float*)d_in[16];
    const float* fcl2_b = (const float*)d_in[17];

    char* ws = (char*)d_ws;
    const size_t BHNC = (size_t)Bn * Hn * Nn * Cn;      // 8388608
    _Float16* Qh  = (_Float16*)(ws);
    _Float16* Kh  = (_Float16*)(ws + BHNC * 2);
    _Float16* Vt  = (_Float16*)(ws + BHNC * 4);
    float*    Ohd = (float*)   (ws + BHNC * 6);

    float* out_y = (float*)d_out;
    float* out_p = out_y + (size_t)Bn * Nn * Cn;

    prep_kernel<<<dim3(Bn * (Nn / 32)), dim3(128), 0, stream>>>(
        x, p, fcl1_w, fcl1_b, hq_w, hq_b, hk_w, hk_b, hv_w, hv_b,
        pe1_w, pe1_b, pe2_w, pe2_b, Qh, Kh, Vt);

    attn_kernel<<<dim3(Bn * Hn * (Nn / 128)), dim3(256), 0, stream>>>(
        Qh, Kh, Vt, Ohd);

    post_kernel<<<dim3(Bn * (Nn / 16)), dim3(128), 0, stream>>>(
        Ohd, mh_w, mh_b, fcl2_w, fcl2_b, x, p, out_y, out_p);
}
